// Combination_ANN_17051020165212
// MI455X (gfx1250) — matrix operand order repaired and checked by reading, not compiled, not run on hardware
//
#include <hip/hip_runtime.h>
#include <hip/hip_bf16.h>

typedef __attribute__((ext_vector_type(16))) _Float16 v16h;
typedef __attribute__((ext_vector_type(8)))  float    v8f;
typedef __attribute__((ext_vector_type(4)))  float    vf4;

#define NWAVES   4
#define LSTRIDE  36   // words per LDS row: 32 cols max + pad so lane-halves hit disjoint banks

static __device__ __forceinline__ v8f wmma16(v16h a, v16h b, v8f c) {
  // D = A(16x32 f16) x B(32x16 f16) + C(16x16 f32)
  return __builtin_amdgcn_wmma_f32_16x16x32_f16(false, a, false, b, (short)0, c, false, false);
}

// One block per (s, sfp): sfp==0 -> plain observations, sfp in 1..250 -> shuffle sf=sfp-1.
// Each wave processes 16-row tiles of the 400 rows independently.
__global__ __launch_bounds__(128) void comb_ann_kernel(
    const float* __restrict__ obs,   // (16,400,16)
    const float* __restrict__ mu,    // (16,)
    const float* __restrict__ Sm,    // (16,16)  Sigma^{-1/2}
    const int*   __restrict__ perm,  // (250,4,16,400)
    const float* __restrict__ W1,    // (16,32)
    const float* __restrict__ b1,    // (32,)
    const float* __restrict__ W2,    // (32,16)
    const float* __restrict__ b2,    // (16,)
    const float* __restrict__ W3,    // (16,1)
    const float* __restrict__ b3,    // (1,)
    float* __restrict__ out)         // (16,100400)
{
  __shared__ __align__(16) float lT[NWAVES][16 * LSTRIDE];
  const int lane = threadIdx.x & 31;
  const int wave = threadIdx.x >> 5;
  const int n    = lane & 15;    // column / row-in-tile id
  const int hi   = lane >> 4;    // lane half
  float* T = lT[wave];           // wave-private transpose buffer

  const int s   = blockIdx.x / 251;
  const int sfp = blockIdx.x % 251;

  // ---- per-wave constants in WMMA B layout: lane holds col n; elem e => K = 8*hi + e + (e & 8) ----
  v16h bSm = {}, bW1a = {}, bW1b = {}, bW2 = {};
  #pragma unroll
  for (int e = 0; e < 16; ++e) {
    const int k = 8 * hi + e + (e & 8);
    if (e < 8) {                                    // K=16..31 (e>=8) multiplies zero-padded A elems
      bSm[e]  = (_Float16)Sm[n * 16 + k];           // B[k][n] = Sm^T[k][n] = Sm[n][k]
      bW1a[e] = (_Float16)W1[k * 32 + n];
      bW1b[e] = (_Float16)W1[k * 32 + 16 + n];
    }
    bW2[e] = (_Float16)W2[k * 16 + n];              // full K=32
  }
  // fold mu into stage-1 accumulator: c0[i] = -sum_j Sm[i,j]*mu[j]
  float c0v = 0.f;
  #pragma unroll
  for (int j = 0; j < 16; ++j) c0v -= Sm[n * 16 + j] * mu[j];
  const float b1a = b1[n], b1b = b1[n + 16], b2v = b2[n], b3s = b3[0];
  const vf4 w3a = ((const vf4*)W3)[2 * hi];
  const vf4 w3b = ((const vf4*)W3)[2 * hi + 1];

  const int g0 = 2 * hi, g1 = g0 + 1;               // this lane's two feature groups
  const int sfm = (sfp > 0) ? (sfp - 1) : 0;
  const float* obsS = obs + (size_t)s * 6400;
  const int* p0 = perm + ((size_t)(sfm * 4 + g0) * 16 + s) * 400;
  const int* p1 = perm + ((size_t)(sfm * 4 + g1) * 16 + s) * 400;
  float* outRow = out + (size_t)s * 100400 + (size_t)sfp * 400;

  for (int tile = wave; tile < 25; tile += NWAVES) {
    const int t = tile * 16 + n;

    // ---- gather this lane's half-row (A layout: row n, features hi*8 .. hi*8+7) ----
    int pt0 = t, pt1 = t;
    if (sfp > 0) { pt0 = p0[t]; pt1 = p1[t]; }
    const vf4 f0 = *(const vf4*)(obsS + (size_t)pt0 * 16 + g0 * 4);
    const vf4 f1 = *(const vf4*)(obsS + (size_t)pt1 * 16 + g1 * 4);

    v16h a1 = {};                                    // K=16..31 stay zero
    a1[0]=(_Float16)f0.x; a1[1]=(_Float16)f0.y; a1[2]=(_Float16)f0.z; a1[3]=(_Float16)f0.w;
    a1[4]=(_Float16)f1.x; a1[5]=(_Float16)f1.y; a1[6]=(_Float16)f1.z; a1[7]=(_Float16)f1.w;

    // ---- stage 1: Y = X @ Sm^T + c0 ----
    v8f acc = {c0v, c0v, c0v, c0v, c0v, c0v, c0v, c0v};
    acc = wmma16(a1, bSm, acc);

    // D layout (lane=col n, VGPR v=row v+8*hi) -> row layout via wave-private LDS
    #pragma unroll
    for (int v = 0; v < 8; ++v) T[(v + 8 * hi) * LSTRIDE + n] = acc[v];
    asm volatile("" ::: "memory");                   // DS unit is in-order per wave
    const vf4 y0 = *(const vf4*)&T[n * LSTRIDE + hi * 8];
    const vf4 y1 = *(const vf4*)&T[n * LSTRIDE + hi * 8 + 4];
    asm volatile("" ::: "memory");
    v16h a2 = {};
    a2[0]=(_Float16)y0.x; a2[1]=(_Float16)y0.y; a2[2]=(_Float16)y0.z; a2[3]=(_Float16)y0.w;
    a2[4]=(_Float16)y1.x; a2[5]=(_Float16)y1.y; a2[6]=(_Float16)y1.z; a2[7]=(_Float16)y1.w;

    // ---- stage 2: H1 = leaky(Y @ W1 + b1), 32 cols = two 16x16 D tiles ----
    v8f h1a = {b1a, b1a, b1a, b1a, b1a, b1a, b1a, b1a};
    v8f h1b = {b1b, b1b, b1b, b1b, b1b, b1b, b1b, b1b};
    h1a = wmma16(a2, bW1a, h1a);
    h1b = wmma16(a2, bW1b, h1b);
    #pragma unroll
    for (int v = 0; v < 8; ++v) {
      float xa = h1a[v]; xa = fmaxf(xa, 0.01f * xa);   // leaky_relu
      float xb = h1b[v]; xb = fmaxf(xb, 0.01f * xb);
      T[(v + 8 * hi) * LSTRIDE + n]      = xa;
      T[(v + 8 * hi) * LSTRIDE + 16 + n] = xb;
    }
    asm volatile("" ::: "memory");
    // A layout for K=32: VGPR0-3 <- cols hi*8..+7, VGPR4-7 <- cols 16+hi*8..+7
    const vf4 x0 = *(const vf4*)&T[n * LSTRIDE + hi * 8];
    const vf4 x1 = *(const vf4*)&T[n * LSTRIDE + hi * 8 + 4];
    const vf4 x2 = *(const vf4*)&T[n * LSTRIDE + 16 + hi * 8];
    const vf4 x3 = *(const vf4*)&T[n * LSTRIDE + 16 + hi * 8 + 4];
    asm volatile("" ::: "memory");
    v16h a3;
    a3[0]=(_Float16)x0.x;  a3[1]=(_Float16)x0.y;  a3[2]=(_Float16)x0.z;  a3[3]=(_Float16)x0.w;
    a3[4]=(_Float16)x1.x;  a3[5]=(_Float16)x1.y;  a3[6]=(_Float16)x1.z;  a3[7]=(_Float16)x1.w;
    a3[8]=(_Float16)x2.x;  a3[9]=(_Float16)x2.y;  a3[10]=(_Float16)x2.z; a3[11]=(_Float16)x2.w;
    a3[12]=(_Float16)x3.x; a3[13]=(_Float16)x3.y; a3[14]=(_Float16)x3.z; a3[15]=(_Float16)x3.w;

    // ---- stage 3: H2 = leaky(H1 @ W2 + b2), exact K=32 ----
    v8f h2 = {b2v, b2v, b2v, b2v, b2v, b2v, b2v, b2v};
    h2 = wmma16(a3, bW2, h2);
    #pragma unroll
    for (int v = 0; v < 8; ++v) {
      float x = h2[v]; x = fmaxf(x, 0.01f * x);
      T[(v + 8 * hi) * LSTRIDE + n] = x;
    }
    asm volatile("" ::: "memory");

    // ---- stage 4: out = sigmoid(H2 @ W3 + b3) : per-lane half-dot + cross-half combine ----
    const vf4 z0 = *(const vf4*)&T[n * LSTRIDE + hi * 8];
    const vf4 z1 = *(const vf4*)&T[n * LSTRIDE + hi * 8 + 4];
    asm volatile("" ::: "memory");
    float part = z0.x * w3a.x + z0.y * w3a.y + z0.z * w3a.z + z0.w * w3a.w
               + z1.x * w3b.x + z1.y * w3b.y + z1.z * w3b.z + z1.w * w3b.w;
    part += __shfl_xor(part, 16, 32);                 // combine the two feature halves
    const float sg = 1.f / (1.f + __expf(-(part + b3s)));
    if (hi == 0) outRow[t] = sg;                      // coalesced 16-lane store
  }
}

extern "C" void kernel_launch(void* const* d_in, const int* in_sizes, int n_in,
                              void* d_out, int out_size, void* d_ws, size_t ws_size,
                              hipStream_t stream) {
  (void)in_sizes; (void)n_in; (void)out_size; (void)d_ws; (void)ws_size;
  const float* obs  = (const float*)d_in[0];
  const float* mu   = (const float*)d_in[1];
  const float* Sm   = (const float*)d_in[2];
  const int*   perm = (const int*)d_in[3];
  const float* W1   = (const float*)d_in[4];
  const float* b1   = (const float*)d_in[5];
  const float* W2   = (const float*)d_in[6];
  const float* b2   = (const float*)d_in[7];
  const float* W3   = (const float*)d_in[8];
  const float* b3   = (const float*)d_in[9];
  float* out = (float*)d_out;

  dim3 grid(16 * 251);   // (s, sfp) pairs: 16 * (1 + 250)
  dim3 block(128);       // 4 wave32 waves, each owns independent 16-row tiles
  hipLaunchKernelGGL(comb_ann_kernel, grid, block, 0, stream,
                     obs, mu, Sm, perm, W1, b1, W2, b2, W3, b3, out);
}
